// GlobalAttentionBlock_137438953873
// MI455X (gfx1250) — matrix operand order repaired and checked by reading, not compiled, not run on hardware
//
#include <hip/hip_runtime.h>
#include <hip/hip_bf16.h>

typedef __attribute__((ext_vector_type(16))) __bf16 bf16x16;
typedef __attribute__((ext_vector_type(8)))  float  f32x8;
typedef __attribute__((ext_vector_type(4)))  unsigned int u32x4;
typedef __attribute__((ext_vector_type(8)))  int i32x8;
typedef __attribute__((ext_vector_type(4)))  int i32x4;

constexpr int D  = 768;
constexpr int H  = 12;
constexpr int DH = 64;
constexpr int T  = 8192;
constexpr int B  = 4;
constexpr int S  = 16;          // stride of global rows
constexpr int G  = T / S;       // 512 global rows per (b,h)
constexpr int F  = 3072;        // MLP dim
constexpr int ROWS = B * T;     // 32768

// ---------------------------------------------------------------------------
// WMMA helpers (CDNA5 16x16x32 bf16, f32 accumulate)
// ---------------------------------------------------------------------------
__device__ __forceinline__ f32x8 wmma_bf16(bf16x16 a, bf16x16 b, f32x8 c) {
  return __builtin_amdgcn_wmma_f32_16x16x32_bf16(false, a, false, b,
                                                 (short)0, c, false, false);
}

__device__ __forceinline__ f32x8 zero8() {
  f32x8 z;
#pragma unroll
  for (int i = 0; i < 8; ++i) z[i] = 0.0f;
  return z;
}

// A fragment: 16(M) x 32(K) bf16 from a row-major buffer with leading dim ld.
// lanes 0-15 row M=lane, elems 0-15 -> K0-15;
// lanes 16-31 same rows, K16-31.
__device__ __forceinline__ bf16x16 load_a_frag(const __bf16* p, int ld) {
  const int lane = threadIdx.x & 31;
  const int row  = lane & 15;
  const int ko   = (lane >> 4) << 4;  // 0 or 16
  const __bf16* r = p + row * ld;
  bf16x16 a;
#pragma unroll
  for (int e = 0; e < 16; ++e) a[e] = r[ko + e];
  return a;
}

// B fragment: 32(K) x 16(N) bf16 from K-major [K][N] storage (ld = row stride).
// lanes 0-15: column N=lane, K0-15; lanes 16-31: column N=lane-16, K16-31.
__device__ __forceinline__ bf16x16 load_b_frag(const __bf16* p, int ld) {
  const int lane = threadIdx.x & 31;
  const int n  = lane & 15;
  const int ko = (lane >> 4) << 4;  // 0 or 16
  bf16x16 b;
#pragma unroll
  for (int e = 0; e < 16; ++e) b[e] = p[(ko + e) * ld + n];
  return b;
}

// B fragment from transposed (N-major) storage: element (k,n) at p[n*ld + k].
__device__ __forceinline__ bf16x16 load_b_frag_t(const __bf16* p, int ld) {
  const int lane = threadIdx.x & 31;
  const int n  = lane & 15;
  const int ko = (lane >> 4) << 4;
  const __bf16* r = p + n * ld;
  bf16x16 b;
#pragma unroll
  for (int e = 0; e < 16; ++e) b[e] = r[ko + e];
  return b;
}

// ---------------------------------------------------------------------------
// Tensor Data Mover: 2D tile (tile_d0 elems/row x tile_d1 rows) of bf16 from
// global (row stride = stride_elems) into contiguous LDS at lds_byte_off.
// D# layout per CDNA5 ISA 8.3/8.4. Tensor declared == tile (always in-bounds;
// stride may exceed dim per spec). data_size code 1 = 2 bytes.
// Toolchain uses the 6-arg builtin: (g0, g1, g2, g3, extra_g, cpol).
// ---------------------------------------------------------------------------
__device__ __forceinline__ unsigned lds_byte_off(const void* p) {
  // Generic LDS addresses carry the in-group byte offset in the low 32 bits
  // (ISA 10.2).
  return (unsigned)(unsigned long long)p;
}

__device__ __forceinline__ void tdm_load_tile_bf16(
    unsigned lds_off, const __bf16* gptr,
    unsigned tile_d0, unsigned tile_d1, unsigned stride_elems) {
  const unsigned long long ga = (unsigned long long)gptr;
  u32x4 g0;
  g0[0] = 1u;                                   // count=1, user desc
  g0[1] = lds_off;                              // lds_addr (bytes)
  g0[2] = (unsigned)(ga & 0xffffffffu);         // global_addr[31:0]
  g0[3] = (unsigned)((ga >> 32) & 0x01ffffffu)  // global_addr[56:32]
        | 0x80000000u;                          // type=2 ("image")
  i32x8 g1;
  g1[0] = 0x00010000;                           // data_size=1 (2B), no mcast
  g1[1] = (int)((tile_d0 & 0xffffu) << 16);     // tensor_dim0[15:0]
  g1[2] = (int)((tile_d0 >> 16) | ((tile_d1 & 0xffffu) << 16)); // td0 hi, td1 lo
  g1[3] = (int)((tile_d1 >> 16) | ((tile_d0 & 0xffffu) << 16)); // td1 hi, tile_dim0
  g1[4] = (int)(tile_d1 & 0xffffu);             // tile_dim1 (tile_dim2 = 0)
  g1[5] = (int)stride_elems;                    // tensor_dim0_stride[31:0]
  g1[6] = 0;                                    // stride hi / dim1_stride lo
  g1[7] = 0;
  i32x4 g2 = {0, 0, 0, 0};
  i32x4 g3 = {0, 0, 0, 0};
  i32x8 gx = {0, 0, 0, 0, 0, 0, 0, 0};
  __builtin_amdgcn_tensor_load_to_lds(g0, g1, g2, g3, gx, 0);
}

// ---------------------------------------------------------------------------
// Elementwise fp32 -> bf16 conversion (weights)
// ---------------------------------------------------------------------------
__global__ __launch_bounds__(256) void f32_to_bf16_kernel(
    const float* __restrict__ in, __bf16* __restrict__ out, int n) {
  int i = blockIdx.x * 256 + threadIdx.x;
  if (i < n) out[i] = (__bf16)in[i];
}

// ---------------------------------------------------------------------------
// LayerNorm: one block per row (768 elems), output bf16
// ---------------------------------------------------------------------------
__global__ __launch_bounds__(256) void layernorm_kernel(
    const float* __restrict__ x, const float* __restrict__ w,
    const float* __restrict__ b, __bf16* __restrict__ out) {
  __shared__ float s1[256];
  __shared__ float s2[256];
  const int tid = threadIdx.x;
  const size_t row = blockIdx.x;
  const float* xr = x + row * D;
  float v0 = xr[tid], v1 = xr[tid + 256], v2 = xr[tid + 512];
  s1[tid] = v0 + v1 + v2;
  s2[tid] = v0 * v0 + v1 * v1 + v2 * v2;
  __syncthreads();
#pragma unroll
  for (int o = 128; o > 0; o >>= 1) {
    if (tid < o) { s1[tid] += s1[tid + o]; s2[tid] += s2[tid + o]; }
    __syncthreads();
  }
  const float mu  = s1[0] * (1.0f / D);
  const float var = s2[0] * (1.0f / D) - mu * mu;
  const float rs  = rsqrtf(var + 1e-5f);
  __bf16* orow = out + row * D;
  orow[tid]       = (__bf16)((v0 - mu) * rs * w[tid]       + b[tid]);
  orow[tid + 256] = (__bf16)((v1 - mu) * rs * w[tid + 256] + b[tid + 256]);
  orow[tid + 512] = (__bf16)((v2 - mu) * rs * w[tid + 512] + b[tid + 512]);
}

// ---------------------------------------------------------------------------
// Generic WMMA GEMM: C[M,N] = A[M,K](bf16) * W[N,K]^T(bf16) + bias.
// Block = 256 threads (8 waves, 2x4), block tile 128x256, wave tile 64x64
// (16 WMMAs per K-step per wave). TDM (tensor_load_to_lds) stages A/W tiles
// into double-buffered LDS; wave 0 issues the DMA and gates on TENSORcnt.
// Epilogues:
//   EPI 0: QKV scatter (q scaled 1/8 -> Qg global rows; k,v head-major;
//          v also seeds O pass-through)
//   EPI 1: f32 residual add:  out_f32 = resid + C
//   EPI 2: exact GELU -> bf16
// ---------------------------------------------------------------------------
template <int EPI>
__global__ __launch_bounds__(256) void gemm_bf16_kernel(
    const __bf16* __restrict__ A, const __bf16* __restrict__ W,
    const float* __restrict__ bias, int M, int N, int K,
    float* __restrict__ out_f32, const float* __restrict__ resid,
    __bf16* __restrict__ out_bf16,
    __bf16* __restrict__ qg, __bf16* __restrict__ kk,
    __bf16* __restrict__ vv, __bf16* __restrict__ ob) {
  __shared__ __attribute__((aligned(16))) __bf16 As[2][128 * 32];  // 16 KB
  __shared__ __attribute__((aligned(16))) __bf16 Bs[2][256 * 32];  // 32 KB
  const int tid  = threadIdx.x;
  const int lane = tid & 31;
  const int wave = tid >> 5;
  const int wm = wave >> 2;        // 0..1  (64 rows each)
  const int wn = wave & 3;         // 0..3  (64 cols each)
  const int m0 = blockIdx.y * 128;
  const int n0 = blockIdx.x * 256;

  f32x8 acc[4][4];
#pragma unroll
  for (int i = 0; i < 4; ++i)
#pragma unroll
    for (int j = 0; j < 4; ++j) acc[i][j] = zero8();

  const __bf16* Abase = A + (size_t)m0 * K;
  const __bf16* Wbase = W + (size_t)n0 * K;
  const int nk = K / 32;

  if (wave == 0) {   // prologue DMA for k-tile 0
    tdm_load_tile_bf16(lds_byte_off(&As[0][0]), Abase, 32, 128, (unsigned)K);
    tdm_load_tile_bf16(lds_byte_off(&Bs[0][0]), Wbase, 32, 256, (unsigned)K);
  }

  for (int it = 0; it < nk; ++it) {
    const int cur = it & 1;
    if (wave == 0) __builtin_amdgcn_s_wait_tensorcnt((unsigned short)0);
    __syncthreads();   // buffer `cur` ready for everyone; `cur^1` drained
    if (wave == 0 && it + 1 < nk) {
      tdm_load_tile_bf16(lds_byte_off(&As[cur ^ 1][0]),
                         Abase + (it + 1) * 32, 32, 128, (unsigned)K);
      tdm_load_tile_bf16(lds_byte_off(&Bs[cur ^ 1][0]),
                         Wbase + (it + 1) * 32, 32, 256, (unsigned)K);
    }
    bf16x16 af[4], bfr[4];
#pragma unroll
    for (int i = 0; i < 4; ++i)
      af[i] = load_a_frag(&As[cur][(wm * 64 + i * 16) * 32], 32);
#pragma unroll
    for (int j = 0; j < 4; ++j)
      bfr[j] = load_b_frag_t(&Bs[cur][(wn * 64 + j * 16) * 32], 32);
#pragma unroll
    for (int i = 0; i < 4; ++i)
#pragma unroll
      for (int j = 0; j < 4; ++j)
        acc[i][j] = wmma_bf16(af[i], bfr[j], acc[i][j]);
    __syncthreads();   // done reading buffer `cur`
  }

  // Epilogue. C layout: lane half hl in {0,1}: m = 8*hl + r, n = lane&15.
  const int hl = lane >> 4, nl = lane & 15;
#pragma unroll
  for (int i = 0; i < 4; ++i) {
#pragma unroll
    for (int j = 0; j < 4; ++j) {
#pragma unroll
      for (int r = 0; r < 8; ++r) {
        const int m = m0 + wm * 64 + i * 16 + hl * 8 + r;
        const int n = n0 + wn * 64 + j * 16 + nl;
        float v = acc[i][j][r] + bias[n];
        if (EPI == 0) {
          const int sect = n / D, c = n % D;
          const int hh = c >> 6, dd = c & 63;
          const int bb = m / T, tt = m % T;
          const size_t bh = (size_t)(bb * H + hh);
          if (sect == 0) {
            if ((tt & (S - 1)) == 0)
              qg[(bh * G + (tt >> 4)) * DH + dd] = (__bf16)(v * 0.125f);
          } else if (sect == 1) {
            kk[(bh * T + tt) * DH + dd] = (__bf16)v;
          } else {
            vv[(bh * T + tt) * DH + dd] = (__bf16)v;
            ob[(size_t)m * D + c] = (__bf16)v;   // pass-through rows: o = v
          }
        } else if (EPI == 1) {
          out_f32[(size_t)m * N + n] = resid[(size_t)m * N + n] + v;
        } else {
          const float g = 0.5f * v * (1.0f + erff(v * 0.70710678118654752f));
          out_bf16[(size_t)m * N + n] = (__bf16)g;
        }
      }
    }
  }
}

// ---------------------------------------------------------------------------
// Global-row flash attention. One wave owns 16 global query rows; block has
// 4 waves (64 rows); grid = B*H*(G/64) = 384 blocks. Streams 32 keys/iter:
// 4 WMMAs for scores (dh=64 -> 2 K-chunks, 2 key tiles), online softmax,
// P restaged via LDS into A-fragment layout, 4 WMMAs for P*V.
// ---------------------------------------------------------------------------
__global__ __launch_bounds__(128) void attn_global_kernel(
    const __bf16* __restrict__ Qg, const __bf16* __restrict__ Kb,
    const __bf16* __restrict__ Vb, __bf16* __restrict__ Ob) {
  __shared__ __attribute__((aligned(16))) __bf16 Plds[4][16 * 32];
  const int wave = threadIdx.x >> 5;
  const int lane = threadIdx.x & 31;
  const int nchunks = G / 64;                 // 8
  const int gchunk = blockIdx.x % nchunks;
  const int bh = blockIdx.x / nchunks;        // 0..47
  const int bb = bh / H, hh = bh % H;
  const int g0 = gchunk * 64 + wave * 16;

  const __bf16* qbase = Qg + ((size_t)bh * G + g0) * DH;
  const bf16x16 aq0 = load_a_frag(qbase, DH);       // d = 0..31
  const bf16x16 aq1 = load_a_frag(qbase + 32, DH);  // d = 32..63
  const __bf16* Kh = Kb + (size_t)bh * T * DH;
  const __bf16* Vh = Vb + (size_t)bh * T * DH;

  f32x8 acc[4];
  float rmax[8], rsum[8];
#pragma unroll
  for (int j = 0; j < 4; ++j) acc[j] = zero8();
#pragma unroll
  for (int r = 0; r < 8; ++r) { rmax[r] = -1e30f; rsum[r] = 0.0f; }

  const int hl = lane >> 4, nl = lane & 15;
  __bf16* pw = &Plds[wave][0];

  for (int kt = 0; kt < T; kt += 32) {
    // scores for keys [kt, kt+32)
    f32x8 s0 = zero8(), s1 = zero8();
    s0 = wmma_bf16(aq0, load_b_frag_t(Kh + (size_t)kt * DH, DH), s0);
    s0 = wmma_bf16(aq1, load_b_frag_t(Kh + (size_t)kt * DH + 32, DH), s0);
    s1 = wmma_bf16(aq0, load_b_frag_t(Kh + (size_t)(kt + 16) * DH, DH), s1);
    s1 = wmma_bf16(aq1, load_b_frag_t(Kh + (size_t)(kt + 16) * DH + 32, DH), s1);

    // online softmax: row-wise max/sum over the 16-lane half holding the row
    float cm[8], p0[8], p1[8], ps[8];
#pragma unroll
    for (int r = 0; r < 8; ++r) cm[r] = fmaxf(s0[r], s1[r]);
#pragma unroll
    for (int off = 8; off >= 1; off >>= 1)
#pragma unroll
      for (int r = 0; r < 8; ++r)
        cm[r] = fmaxf(cm[r], __shfl_xor(cm[r], off, 32));
#pragma unroll
    for (int r = 0; r < 8; ++r) {
      const float nm = fmaxf(rmax[r], cm[r]);
      const float sc = __expf(rmax[r] - nm);
      rmax[r] = nm;
      rsum[r] *= sc;
      p0[r] = __expf(s0[r] - nm);
      p1[r] = __expf(s1[r] - nm);
      ps[r] = p0[r] + p1[r];
      acc[0][r] *= sc; acc[1][r] *= sc; acc[2][r] *= sc; acc[3][r] *= sc;
    }
#pragma unroll
    for (int off = 8; off >= 1; off >>= 1)
#pragma unroll
      for (int r = 0; r < 8; ++r) ps[r] += __shfl_xor(ps[r], off, 32);
#pragma unroll
    for (int r = 0; r < 8; ++r) rsum[r] += ps[r];

    // restage P (C layout) into row-major 16x32 for the A fragment
#pragma unroll
    for (int r = 0; r < 8; ++r) {
      const int mrow = hl * 8 + r;
      pw[mrow * 32 + nl]      = (__bf16)p0[r];
      pw[mrow * 32 + 16 + nl] = (__bf16)p1[r];
    }
    __syncthreads();
    const bf16x16 pa = load_a_frag(pw, 32);
    __syncthreads();

#pragma unroll
    for (int j = 0; j < 4; ++j) {
      const bf16x16 bv = load_b_frag(Vh + (size_t)kt * DH + j * 16, DH);
      acc[j] = wmma_bf16(pa, bv, acc[j]);
    }
  }

  // normalize + scatter to global rows of O (token = (g0+m)*S)
#pragma unroll
  for (int r = 0; r < 8; ++r) {
    const float inv = 1.0f / rsum[r];
    const int mrow = hl * 8 + r;
    const int tok = (g0 + mrow) * S;
    __bf16* orow = Ob + ((size_t)bb * T + tok) * D + hh * DH;
#pragma unroll
    for (int j = 0; j < 4; ++j)
      orow[j * 16 + nl] = (__bf16)(acc[j][r] * inv);
  }
}

// ---------------------------------------------------------------------------
// Host-side orchestration
// ---------------------------------------------------------------------------
extern "C" void kernel_launch(void* const* d_in, const int* in_sizes, int n_in,
                              void* d_out, int out_size, void* d_ws,
                              size_t ws_size, hipStream_t stream) {
  (void)in_sizes; (void)n_in; (void)out_size; (void)ws_size;
  const float* x     = (const float*)d_in[0];
  const float* in_w  = (const float*)d_in[1];
  const float* in_b  = (const float*)d_in[2];
  const float* out_w = (const float*)d_in[3];
  const float* out_b = (const float*)d_in[4];
  const float* ln1w  = (const float*)d_in[5];
  const float* ln1b  = (const float*)d_in[6];
  const float* ln2w  = (const float*)d_in[7];
  const float* ln2b  = (const float*)d_in[8];
  const float* w1    = (const float*)d_in[9];
  const float* b1    = (const float*)d_in[10];
  const float* w2    = (const float*)d_in[11];
  const float* b2    = (const float*)d_in[12];
  float* out = (float*)d_out;

  char* p = (char*)d_ws;
  auto alloc = [&](size_t bytes) -> void* {
    void* r = (void*)p;
    p += (bytes + 255) & ~(size_t)255;
    return r;
  };
  __bf16* inWb  = (__bf16*)alloc((size_t)3 * D * D * 2);
  __bf16* outWb = (__bf16*)alloc((size_t)D * D * 2);
  __bf16* w1b   = (__bf16*)alloc((size_t)F * D * 2);
  __bf16* w2b   = (__bf16*)alloc((size_t)D * F * 2);
  __bf16* xn    = (__bf16*)alloc((size_t)ROWS * D * 2);   // reused for xn2
  __bf16* QgB   = (__bf16*)alloc((size_t)B * H * G * DH * 2);
  __bf16* KbB   = (__bf16*)alloc((size_t)B * H * T * DH * 2);
  __bf16* VbB   = (__bf16*)alloc((size_t)B * H * T * DH * 2);
  __bf16* ObB   = (__bf16*)alloc((size_t)ROWS * D * 2);
  float*  x1    = (float*)alloc((size_t)ROWS * D * 4);
  __bf16* hB    = (__bf16*)alloc((size_t)ROWS * F * 2);

  auto cvt = [&](const float* src, __bf16* dst, int n) {
    f32_to_bf16_kernel<<<(n + 255) / 256, 256, 0, stream>>>(src, dst, n);
  };
  cvt(in_w, inWb, 3 * D * D);
  cvt(out_w, outWb, D * D);
  cvt(w1, w1b, F * D);
  cvt(w2, w2b, D * F);

  // LN1 -> xn (bf16)
  layernorm_kernel<<<ROWS, 256, 0, stream>>>(x, ln1w, ln1b, xn);

  // QKV projection with scatter epilogue
  gemm_bf16_kernel<0><<<dim3(3 * D / 256, ROWS / 128), 256, 0, stream>>>(
      xn, inWb, in_b, ROWS, 3 * D, D,
      nullptr, nullptr, nullptr, QgB, KbB, VbB, ObB);

  // global-row attention overwrites stride-16 rows of O
  attn_global_kernel<<<B * H * (G / 64), 128, 0, stream>>>(QgB, KbB, VbB, ObB);

  // out-proj + residual -> x1 (f32)
  gemm_bf16_kernel<1><<<dim3(D / 256, ROWS / 128), 256, 0, stream>>>(
      ObB, outWb, out_b, ROWS, D, D,
      x1, x, nullptr, nullptr, nullptr, nullptr, nullptr);

  // LN2 -> xn (bf16, reused)
  layernorm_kernel<<<ROWS, 256, 0, stream>>>(x1, ln2w, ln2b, xn);

  // MLP up + exact GELU -> h (bf16)
  gemm_bf16_kernel<2><<<dim3(F / 256, ROWS / 128), 256, 0, stream>>>(
      xn, w1b, b1, ROWS, F, D,
      nullptr, nullptr, hB, nullptr, nullptr, nullptr, nullptr);

  // MLP down + residual -> out (f32)
  gemm_bf16_kernel<1><<<dim3(D / 256, ROWS / 128), 256, 0, stream>>>(
      hB, w2b, b2, ROWS, D, F,
      out, x1, nullptr, nullptr, nullptr, nullptr, nullptr);
}
